// Reinforce_84275848282299
// MI455X (gfx1250) — compile-verified, matrix operand order checked by reading, not run on hardware
//
#include <hip/hip_runtime.h>
#include <hip/hip_bf16.h>

// ---------------------------------------------------------------------------
// Problem dims
// ---------------------------------------------------------------------------
#define VOCAB  10000
#define SEQ    128
#define BSZ    64
#define H      650
#define E      650
// padded dims
#define KP     672     // 650 padded to 21*32 (WMMA K granularity)
#define HPG    656     // per-gate padded width (41*16)
#define GP     2624    // 4*HPG, gate-major padded (164*16 = 82*32)
#define NTOK   8192    // SEQ*BSZ
#define VROWS_P 10048  // vocab rows padded to 628*16 (628 = 157*4 n-tiles)
#define NGROUPS 157    // 628 n-tiles / 4 per group

typedef __attribute__((ext_vector_type(16))) __bf16 bf16x16;
typedef __attribute__((ext_vector_type(8)))  __bf16 bf16x8;
typedef __attribute__((ext_vector_type(8)))  float  v8f;

union Frag { bf16x16 v; bf16x8 h[2]; };

__device__ inline __bf16 f2bf(float f) {
  unsigned u = __builtin_bit_cast(unsigned, f);
  unsigned r = (u + 0x7FFFu + ((u >> 16) & 1u)) >> 16;
  unsigned short s = (unsigned short)r;
  return __builtin_bit_cast(__bf16, s);
}
__device__ inline float sigmoidf(float x) { return 1.0f / (1.0f + __expf(-x)); }

__device__ inline Frag load_frag(const __bf16* p) {   // p = row base + half*8
  Frag f;
  f.h[0] = *reinterpret_cast<const bf16x8*>(p);
  f.h[1] = *reinterpret_cast<const bf16x8*>(p + 16);
  return f;
}

// ---------------------------------------------------------------------------
// Weight conversion / padding kernels (f32 -> bf16, zero-padded)
// ---------------------------------------------------------------------------
__global__ void padcopy_kernel(const float* __restrict__ src, __bf16* __restrict__ dst,
                               int src_rows, int dst_rows) {
  size_t idx = (size_t)blockIdx.x * blockDim.x + threadIdx.x;
  size_t total = (size_t)dst_rows * KP;
  if (idx >= total) return;
  int r = (int)(idx / KP), k = (int)(idx % KP);
  float v = (r < src_rows && k < H) ? src[(size_t)r * H + k] : 0.0f;
  dst[idx] = f2bf(v);
}

__global__ void gatepad_kernel(const float* __restrict__ src, __bf16* __restrict__ dst) {
  // src [4H][650] gate-major -> dst [GP][KP], per-gate row padding
  size_t idx = (size_t)blockIdx.x * blockDim.x + threadIdx.x;
  if (idx >= (size_t)GP * KP) return;
  int n = (int)(idx / KP), k = (int)(idx % KP);
  int g = n / HPG, j = n % HPG;
  float v = (j < H && k < H) ? src[(size_t)(g * H + j) * H + k] : 0.0f;
  dst[idx] = f2bf(v);
}

__global__ void init_states_kernel(const float* __restrict__ h0, const float* __restrict__ c0,
                                   const float* __restrict__ bh0, const float* __restrict__ bc0,
                                   __bf16* __restrict__ h_state, float* __restrict__ c_state) {
  int idx = blockIdx.x * blockDim.x + threadIdx.x;
  if (idx >= 128 * KP) return;
  int rowg = idx / KP, j = idx % KP;
  int b = rowg & 63;
  bool ep = rowg >= 64;
  float h = 0.0f, c = 0.0f;
  if (j < H) {
    h = ep ? h0[b * H + j] : bh0[b * H + j];
    c = ep ? c0[b * H + j] : bc0[b * H + j];
  }
  h_state[idx] = f2bf(h);
  c_state[idx] = c;
}

// ---------------------------------------------------------------------------
// x_proj = emb[tokens] @ W_ih^T + (b_ih + b_hh)   -- shared by both passes
// 2x2 register blocking: wave computes 32x32 output. grid (256, 41), block 64.
// ---------------------------------------------------------------------------
__global__ void xproj_gemm_kernel(const __bf16* __restrict__ embW,
                                  const __bf16* __restrict__ Wihb,
                                  const int* __restrict__ inputs,
                                  const float* __restrict__ b_ih,
                                  const float* __restrict__ b_hh,
                                  float* __restrict__ x_proj) {
  int lane = threadIdx.x & 31, wv = threadIdx.x >> 5;
  int npair = blockIdx.y * 2 + wv;            // < 82
  int m0 = blockIdx.x * 32;                   // < 8192
  int n0 = npair * 32;
  int r = lane & 15, half = lane >> 4;

  int tok0 = inputs[m0 + r];
  int tok1 = inputs[m0 + 16 + r];
  const __bf16* arow0 = embW + (size_t)tok0 * KP + half * 8;
  const __bf16* arow1 = embW + (size_t)tok1 * KP + half * 8;
  const __bf16* brow0 = Wihb + (size_t)(n0 + r) * KP + half * 8;
  const __bf16* brow1 = Wihb + (size_t)(n0 + 16 + r) * KP + half * 8;

  v8f acc00 = {}, acc01 = {}, acc10 = {}, acc11 = {};
  for (int k0 = 0; k0 < KP; k0 += 32) {
    Frag a0 = load_frag(arow0 + k0), a1 = load_frag(arow1 + k0);
    Frag b0 = load_frag(brow0 + k0), b1 = load_frag(brow1 + k0);
    acc00 = __builtin_amdgcn_wmma_f32_16x16x32_bf16(false, a0.v, false, b0.v, (short)0, acc00, false, false);
    acc01 = __builtin_amdgcn_wmma_f32_16x16x32_bf16(false, a0.v, false, b1.v, (short)0, acc01, false, false);
    acc10 = __builtin_amdgcn_wmma_f32_16x16x32_bf16(false, a1.v, false, b0.v, (short)0, acc10, false, false);
    acc11 = __builtin_amdgcn_wmma_f32_16x16x32_bf16(false, a1.v, false, b1.v, (short)0, acc11, false, false);
  }
  int col0 = n0 + (lane & 15), col1 = col0 + 16;
  int g0 = col0 / HPG, j0 = col0 % HPG;
  int g1 = col1 / HPG, j1 = col1 % HPG;
  float bias0 = (j0 < H) ? (b_ih[g0 * H + j0] + b_hh[g0 * H + j0]) : 0.0f;
  float bias1 = (j1 < H) ? (b_ih[g1 * H + j1] + b_hh[g1 * H + j1]) : 0.0f;
  int mr = m0 + (lane >> 4) * 8;
  for (int v = 0; v < 8; ++v) {
    x_proj[(size_t)(mr + v) * GP + col0]      = acc00[v] + bias0;
    x_proj[(size_t)(mr + v) * GP + col1]      = acc01[v] + bias1;
    x_proj[(size_t)(mr + 16 + v) * GP + col0] = acc10[v] + bias0;
    x_proj[(size_t)(mr + 16 + v) * GP + col1] = acc11[v] + bias1;
  }
}

// ---------------------------------------------------------------------------
// Per-timestep recurrent GEMM: gates = x_proj[t] + h_state @ W_hh^T
// Rows 0..63 = base pass, 64..127 = episode pass.
// 2x2 register blocking. grid (4, 41), block 64.
// ---------------------------------------------------------------------------
__global__ void step_gemm_kernel(const __bf16* __restrict__ h_state,
                                 const __bf16* __restrict__ Whhb,
                                 const float* __restrict__ x_proj,
                                 float* __restrict__ gates, int t) {
  int lane = threadIdx.x & 31, wv = threadIdx.x >> 5;
  int npair = blockIdx.y * 2 + wv;            // < 82
  int m0 = blockIdx.x * 32;                   // < 128
  int n0 = npair * 32;
  int r = lane & 15, half = lane >> 4;

  const __bf16* arow0 = h_state + (size_t)(m0 + r) * KP + half * 8;
  const __bf16* arow1 = h_state + (size_t)(m0 + 16 + r) * KP + half * 8;
  const __bf16* brow0 = Whhb + (size_t)(n0 + r) * KP + half * 8;
  const __bf16* brow1 = Whhb + (size_t)(n0 + 16 + r) * KP + half * 8;

  int col0 = n0 + (lane & 15), col1 = col0 + 16;
  int mr = m0 + (lane >> 4) * 8;
  v8f acc00, acc01, acc10, acc11;
  for (int v = 0; v < 8; ++v) {
    int bA = (mr + v) & 63, bB = (mr + 16 + v) & 63;
    acc00[v] = x_proj[(size_t)(t * 64 + bA) * GP + col0];
    acc01[v] = x_proj[(size_t)(t * 64 + bA) * GP + col1];
    acc10[v] = x_proj[(size_t)(t * 64 + bB) * GP + col0];
    acc11[v] = x_proj[(size_t)(t * 64 + bB) * GP + col1];
  }
  for (int k0 = 0; k0 < KP; k0 += 32) {
    Frag a0 = load_frag(arow0 + k0), a1 = load_frag(arow1 + k0);
    Frag b0 = load_frag(brow0 + k0), b1 = load_frag(brow1 + k0);
    acc00 = __builtin_amdgcn_wmma_f32_16x16x32_bf16(false, a0.v, false, b0.v, (short)0, acc00, false, false);
    acc01 = __builtin_amdgcn_wmma_f32_16x16x32_bf16(false, a0.v, false, b1.v, (short)0, acc01, false, false);
    acc10 = __builtin_amdgcn_wmma_f32_16x16x32_bf16(false, a1.v, false, b0.v, (short)0, acc10, false, false);
    acc11 = __builtin_amdgcn_wmma_f32_16x16x32_bf16(false, a1.v, false, b1.v, (short)0, acc11, false, false);
  }
  for (int v = 0; v < 8; ++v) {
    gates[(size_t)(mr + v) * GP + col0]      = acc00[v];
    gates[(size_t)(mr + v) * GP + col1]      = acc01[v];
    gates[(size_t)(mr + 16 + v) * GP + col0] = acc10[v];
    gates[(size_t)(mr + 16 + v) * GP + col1] = acc11[v];
  }
}

// ---------------------------------------------------------------------------
// LSTM cell nonlinearity + noise injection (episode rows only)
// ---------------------------------------------------------------------------
__global__ void lstm_cell_kernel(const float* __restrict__ gates,
                                 const float* __restrict__ noise,
                                 float* __restrict__ c_state,
                                 __bf16* __restrict__ h_state,
                                 __bf16* __restrict__ hout, int t) {
  int idx = blockIdx.x * blockDim.x + threadIdx.x;
  if (idx >= 128 * H) return;
  int rowg = idx / H, j = idx % H;
  int b = rowg & 63;
  bool ep = rowg >= 64;
  const float* gr = gates + (size_t)rowg * GP;
  float gi = gr[j], gf = gr[HPG + j], gg = gr[2 * HPG + j], go = gr[3 * HPG + j];
  float cp = c_state[rowg * KP + j];
  float c2 = sigmoidf(gf) * cp + sigmoidf(gi) * tanhf(gg);
  float h2 = sigmoidf(go) * tanhf(c2);
  size_t tok = (size_t)(t * 64 + b);
  size_t orow = ep ? tok : (size_t)(NTOK) + tok;
  hout[orow * KP + j] = f2bf(h2);
  float hn = h2, cn = c2;
  if (ep) {
    hn = h2 + noise[tok * (2 * H) + j];         // SIGMA = 1
    cn = c2 + noise[tok * (2 * H) + H + j];
  }
  h_state[rowg * KP + j] = f2bf(hn);
  c_state[rowg * KP + j] = cn;
}

// ---------------------------------------------------------------------------
// Fused decoder GEMM + online logsumexp + target-logit pick -> per-token NLL
// One workgroup (8 waves) per 16-row tile; waves sweep 157 groups of 4 vocab
// tiles, reusing the LDS-staged A fragment across the 4 WMMAs of each group.
// Software-prefetch of the next group's B rows (stride 512*KP elements is far
// beyond any HW streamer) -> global_prefetch_b8 hides L2->L0 latency.
// ---------------------------------------------------------------------------
__global__ void decoder_nll_kernel(const __bf16* __restrict__ hout,
                                   const __bf16* __restrict__ Wdecb,
                                   const float* __restrict__ b_dec,
                                   const int* __restrict__ targets,
                                   float* __restrict__ rewards) {
  __shared__ __bf16 As[16 * KP];
  __shared__ float red_mx[8][16], red_sm[8][16];
  __shared__ float tgt_logit[16];
  __shared__ int tgt_col[16];
  int tid = threadIdx.x, lane = tid & 31, wv = tid >> 5;
  int mtile = blockIdx.x;                      // < 1024

  // Stage A tile in LDS (16 consecutive rows of hout are contiguous)
  const unsigned int* src =
      reinterpret_cast<const unsigned int*>(hout + (size_t)mtile * 16 * KP);
  unsigned int* dst = reinterpret_cast<unsigned int*>(As);
  for (int i = tid; i < 16 * (KP / 2); i += 256) dst[i] = src[i];
  // zero the K padding (cols 650..671 -> dwords 325..335 of each 336-dword row)
  for (int i = tid; i < 16 * 11; i += 256) {
    int rr = i / 11, cc = 325 + i % 11;
    dst[rr * (KP / 2) + cc] = 0u;
  }
  if (tid < 16) {
    int row = mtile * 16 + tid;
    tgt_col[tid] = targets[row & (NTOK - 1)];  // base rows reuse same targets
    tgt_logit[tid] = 0.0f;
  }
  __syncthreads();

  int r = lane & 15, half = lane >> 4, mh = half * 8;
  const __bf16* arow = As + r * KP + half * 8;
  float mx[8], sm[8];
  for (int v = 0; v < 8; ++v) { mx[v] = -1e30f; sm[v] = 0.0f; }

  for (int ng = wv; ng < NGROUPS; ng += 8) {
    int nt0 = ng * 4;                          // first of 4 n-tiles
    const __bf16* brow0 = Wdecb + (size_t)(nt0 * 16 + r) * KP + half * 8;
    // prefetch this wave's NEXT group (ng+8 -> +32 tiles -> +512 rows);
    // speculative prefetch: OOB addresses are silently dropped (ISA 10.5)
    __builtin_prefetch(brow0 + (size_t)512 * KP, 0, 1);
    __builtin_prefetch(brow0 + (size_t)(512 + 16) * KP, 0, 1);
    __builtin_prefetch(brow0 + (size_t)(512 + 32) * KP, 0, 1);
    __builtin_prefetch(brow0 + (size_t)(512 + 48) * KP, 0, 1);
    v8f acc0 = {}, acc1 = {}, acc2 = {}, acc3 = {};
    for (int k0 = 0; k0 < KP; k0 += 32) {
      Frag a = load_frag(arow + k0);           // 2 ds_read_b128, reused 4x
      Frag b0 = load_frag(brow0 + k0);
      Frag b1 = load_frag(brow0 + 16 * KP + k0);
      Frag b2 = load_frag(brow0 + 32 * KP + k0);
      Frag b3 = load_frag(brow0 + 48 * KP + k0);
      acc0 = __builtin_amdgcn_wmma_f32_16x16x32_bf16(false, a.v, false, b0.v, (short)0, acc0, false, false);
      acc1 = __builtin_amdgcn_wmma_f32_16x16x32_bf16(false, a.v, false, b1.v, (short)0, acc1, false, false);
      acc2 = __builtin_amdgcn_wmma_f32_16x16x32_bf16(false, a.v, false, b2.v, (short)0, acc2, false, false);
      acc3 = __builtin_amdgcn_wmma_f32_16x16x32_bf16(false, a.v, false, b3.v, (short)0, acc3, false, false);
    }
    v8f* accs[4] = { &acc0, &acc1, &acc2, &acc3 };
    for (int jt = 0; jt < 4; ++jt) {
      int n = (nt0 + jt) * 16 + (lane & 15);
      if (n < VOCAB) {                          // pad tiles are whole-tile skips
        float bd = b_dec[n];
        v8f& A = *accs[jt];
        for (int v = 0; v < 8; ++v) {
          float x = A[v] + bd;
          if (tgt_col[mh + v] == n) tgt_logit[mh + v] = x;  // unique writer/row
          if (x > mx[v]) { sm[v] = sm[v] * __expf(mx[v] - x) + 1.0f; mx[v] = x; }
          else           { sm[v] += __expf(x - mx[v]); }
        }
      }
    }
  }
  // combine across the 16 lanes sharing the same 8 rows (xor stays in half)
  for (int v = 0; v < 8; ++v) {
    for (int off = 8; off; off >>= 1) {
      float omx = __shfl_xor(mx[v], off, 32);
      float osm = __shfl_xor(sm[v], off, 32);
      float M = fmaxf(mx[v], omx);
      sm[v] = sm[v] * __expf(mx[v] - M) + osm * __expf(omx - M);
      mx[v] = M;
    }
  }
  if ((lane & 15) == 0)
    for (int v = 0; v < 8; ++v) { red_mx[wv][mh + v] = mx[v]; red_sm[wv][mh + v] = sm[v]; }
  __syncthreads();
  if (tid < 16) {
    float M = red_mx[0][tid], S = red_sm[0][tid];
    for (int w = 1; w < 8; ++w) {
      float m2 = red_mx[w][tid], s2 = red_sm[w][tid];
      float Mn = fmaxf(M, m2);
      S = S * __expf(M - Mn) + s2 * __expf(m2 - Mn);
      M = Mn;
    }
    rewards[mtile * 16 + tid] = (M + __logf(S)) - tgt_logit[tid];
  }
}

// ---------------------------------------------------------------------------
// logprob[row] = -sum(noise[row]^2)   (SIGMA = 1 -> independent of hc)
// ---------------------------------------------------------------------------
__global__ void logprob_kernel(const float* __restrict__ noise, float* __restrict__ logprob) {
  int row = blockIdx.x;
  int tid = threadIdx.x;
  const float* np = noise + (size_t)row * (2 * H);
  float s = 0.0f;
  for (int k = tid; k < 2 * H; k += 256) { float x = np[k]; s += x * x; }
  for (int off = 16; off; off >>= 1) s += __shfl_xor(s, off, 32);
  __shared__ float wsum[8];
  if ((tid & 31) == 0) wsum[tid >> 5] = s;
  __syncthreads();
  if (tid == 0) {
    float tot = 0.0f;
    for (int w = 0; w < 8; ++w) tot += wsum[w];
    logprob[row] = -tot;
  }
}

// ---------------------------------------------------------------------------
// Backward returns scan + final scalar blend. 1 block, 64 threads (1 per b).
// ---------------------------------------------------------------------------
__global__ void final_kernel(const float* __restrict__ rewards,
                             const float* __restrict__ logprob,
                             const float* __restrict__ alpha,
                             float* __restrict__ out) {
  int b = threadIdx.x;  // 0..63
  float ret = 0.0f, acc_rl = 0.0f, acc_rw = 0.0f, acc_base = 0.0f;
  for (int t = SEQ - 1; t >= 0; --t) {
    float rr = rewards[t * 64 + b];
    ret = rr + ret;                           // GAMMA = 1
    acc_rl += ret * logprob[t * 64 + b];
    acc_rw += rr;
    acc_base += rewards[NTOK + t * 64 + b];
  }
  __shared__ float s_rl[64], s_rw[64], s_b[64];
  s_rl[b] = acc_rl; s_rw[b] = acc_rw; s_b[b] = acc_base;
  __syncthreads();
  if (b == 0) {
    float rl = 0.0f, rw = 0.0f, bb = 0.0f;
    for (int i = 0; i < 64; ++i) { rl += s_rl[i]; rw += s_rw[i]; bb += s_b[i]; }
    float a = alpha[0];
    out[0] = a * (rl / (float)NTOK) + (1.0f - a) * (bb / (float)NTOK);
    out[1] = rw / (float)NTOK;
  }
}

// ---------------------------------------------------------------------------
// Host orchestration
// ---------------------------------------------------------------------------
extern "C" void kernel_launch(void* const* d_in, const int* in_sizes, int n_in,
                              void* d_out, int out_size, void* d_ws, size_t ws_size,
                              hipStream_t stream) {
  const int*   inputs  = (const int*)  d_in[0];
  const int*   targets = (const int*)  d_in[1];
  const float* h0      = (const float*)d_in[2];
  const float* c0      = (const float*)d_in[3];
  const float* bh0     = (const float*)d_in[4];
  const float* bc0     = (const float*)d_in[5];
  const float* alpha   = (const float*)d_in[6];
  const float* noise   = (const float*)d_in[7];
  const float* emb     = (const float*)d_in[8];
  const float* W_ih    = (const float*)d_in[9];
  const float* W_hh    = (const float*)d_in[10];
  const float* b_ih    = (const float*)d_in[11];
  const float* b_hh    = (const float*)d_in[12];
  const float* W_dec   = (const float*)d_in[13];
  const float* b_dec   = (const float*)d_in[14];
  float* out = (float*)d_out;

  // workspace carve-up (all offsets 256B aligned)
  char* ws = (char*)d_ws;
  size_t off = 0;
  auto carve = [&](size_t bytes) -> char* {
    char* p = ws + off;
    off += (bytes + 255) & ~(size_t)255;
    return p;
  };
  __bf16* embW    = (__bf16*)carve((size_t)VOCAB * KP * 2);
  __bf16* Wihb    = (__bf16*)carve((size_t)GP * KP * 2);
  __bf16* Whhb    = (__bf16*)carve((size_t)GP * KP * 2);
  __bf16* Wdecb   = (__bf16*)carve((size_t)VROWS_P * KP * 2);
  float*  x_proj  = (float*) carve((size_t)NTOK * GP * 4);
  float*  gates   = (float*) carve((size_t)128 * GP * 4);
  __bf16* h_state = (__bf16*)carve((size_t)128 * KP * 2);
  float*  c_state = (float*) carve((size_t)128 * KP * 4);
  __bf16* hout    = (__bf16*)carve((size_t)2 * NTOK * KP * 2);
  float*  rewards = (float*) carve((size_t)2 * NTOK * 4);
  float*  logprob = (float*) carve((size_t)NTOK * 4);

  // 1. weight conversion
  {
    size_t n = (size_t)VOCAB * KP;
    padcopy_kernel<<<(int)((n + 255) / 256), 256, 0, stream>>>(emb, embW, VOCAB, VOCAB);
    size_t nd = (size_t)VROWS_P * KP;
    padcopy_kernel<<<(int)((nd + 255) / 256), 256, 0, stream>>>(W_dec, Wdecb, VOCAB, VROWS_P);
    size_t g = (size_t)GP * KP;
    gatepad_kernel<<<(int)((g + 255) / 256), 256, 0, stream>>>(W_ih, Wihb);
    gatepad_kernel<<<(int)((g + 255) / 256), 256, 0, stream>>>(W_hh, Whhb);
  }
  // 2. initial states (base rows 0..63, episode rows 64..127)
  init_states_kernel<<<(128 * KP + 255) / 256, 256, 0, stream>>>(
      h0, c0, bh0, bc0, h_state, c_state);

  // 3. x_proj for every token (shared by both passes), 2x2-blocked WMMA
  xproj_gemm_kernel<<<dim3(NTOK / 32, 41), 64, 0, stream>>>(
      embW, Wihb, inputs, b_ih, b_hh, x_proj);

  // 4. recurrence: 128 sequential steps, both passes batched (M = 128)
  for (int t = 0; t < SEQ; ++t) {
    step_gemm_kernel<<<dim3(4, 41), 64, 0, stream>>>(h_state, Whhb, x_proj, gates, t);
    lstm_cell_kernel<<<(128 * H + 255) / 256, 256, 0, stream>>>(
        gates, noise, c_state, h_state, hout, t);
  }

  // 5. fused decoder + NLL for both streams (rows 0..8191 ep, 8192..16383 base)
  decoder_nll_kernel<<<(2 * NTOK) / 16, 256, 0, stream>>>(
      hout, Wdecb, b_dec, targets, rewards);

  // 6. logprobs from noise
  logprob_kernel<<<NTOK, 256, 0, stream>>>(noise, logprob);

  // 7. returns scan + scalar outputs
  final_kernel<<<1, 64, 0, stream>>>(rewards, logprob, alpha, out);
}
